// MultiHeadAttention_40922448397086
// MI455X (gfx1250) — matrix operand order repaired and checked by reading, not compiled, not run on hardware
//
#include <hip/hip_runtime.h>

// ---------------------------------------------------------------------------
// MHA forward for MI455X (gfx1250, wave32, WMMA + Tensor Data Mover).
// D=512, H=8, DK=64, B=4, S=2048.
// Matmuls: v_wmma_f32_16x16x32_bf16 (f32 accumulate).
// Tile staging: tensor_load_to_lds (TDM) with LDS padding, double-buffered.
// ---------------------------------------------------------------------------

#define DMODEL 512
#define SEQ    2048
#define NHEAD  8
#define DK     64
#define BATCH  4

typedef __attribute__((ext_vector_type(16))) __bf16 v16bf;
typedef __attribute__((ext_vector_type(8)))  float  v8f;
typedef __attribute__((ext_vector_type(4)))  unsigned int u32x4;
typedef __attribute__((ext_vector_type(8)))  int  i32x8;
typedef __attribute__((ext_vector_type(4)))  int  i32x4;

#if defined(__has_builtin)
#if __has_builtin(__builtin_amdgcn_tensor_load_to_lds) && \
    __has_builtin(__builtin_amdgcn_s_wait_tensorcnt)
#define HAVE_TDM 1
#endif
#endif
#ifndef HAVE_TDM
#define HAVE_TDM 0
#endif

__device__ __forceinline__ unsigned short f2bf(float f) {
  unsigned int u = __builtin_bit_cast(unsigned int, f);
  unsigned int rnd = 0x7FFFu + ((u >> 16) & 1u);       // round-to-nearest-even
  return (unsigned short)((u + rnd) >> 16);
}

// Low 32 bits of a flat shared-space pointer = LDS byte offset (ISA 10.2).
__device__ __forceinline__ unsigned lds_addr32(const void* p) {
  return (unsigned)(unsigned long long)p;
}

#if HAVE_TDM
// ---------------------------------------------------------------------------
// TDM: DMA a [tile_h x tile_w] bf16 tile (row stride = stride_elems) from
// global to LDS, inserting pad_amount dwords every 2^(pad_interval+1) dwords
// (ISA 8.3/8.4 descriptor layout). One descriptor, issued by one wave;
// completion tracked with TENSORcnt.
// ---------------------------------------------------------------------------
__device__ __forceinline__ void tdm_load_tile_bf16(
    unsigned long long gaddr, unsigned lds_byte_off,
    int tile_w, int tile_h, int stride_elems,
    int pad_interval_code, int pad_amount_code) {
  u32x4 g0;
  g0.x = 1u;                                            // count=1, user mode
  g0.y = (unsigned)__builtin_amdgcn_readfirstlane((int)lds_byte_off);
  g0.z = (unsigned)__builtin_amdgcn_readfirstlane((int)(unsigned)gaddr);
  g0.w = (unsigned)__builtin_amdgcn_readfirstlane(
             (int)((unsigned)(gaddr >> 32) | (2u << 30)));   // addr[56:32] | type=2
  i32x8 g1;
  g1[0] = (1 << 16) |                                   // data_size = 2 bytes
          (1 << 20) |                                   // pad_enable
          (pad_interval_code << 22) | (pad_amount_code << 25);
  g1[1] = (tile_w & 0xFFFF) << 16;                      // tensor_dim0 lo16
  g1[2] = (tile_w >> 16) | ((tile_h & 0xFFFF) << 16);   // dim0 hi | tensor_dim1 lo
  g1[3] = (tile_h >> 16) | (tile_w << 16);              // dim1 hi | tile_dim0
  g1[4] = tile_h & 0xFFFF;                              // tile_dim1 (tile_dim2=0)
  g1[5] = stride_elems;                                 // tensor_dim0_stride lo32
  g1[6] = 0;                                            // stride hi | dim1_stride lo
  g1[7] = 0;
  i32x4 z4 = {0, 0, 0, 0};
#if __clang_major__ >= 23
  i32x8 z8 = {0, 0, 0, 0, 0, 0, 0, 0};
  __builtin_amdgcn_tensor_load_to_lds(g0, g1, z4, z4, z8, 0);
#else
  __builtin_amdgcn_tensor_load_to_lds(g0, g1, z4, z4, 0);
#endif
}
#endif  // HAVE_TDM

// --- WMMA fragment loaders (wave32 layouts per CDNA5 ISA 7.12.2) -----------
__device__ __forceinline__ v16bf frag_a(const unsigned short* p, int stride) {
  const int lane = threadIdx.x & 31;
  const int half = lane >> 4, r = lane & 15;
  union { unsigned int u[8]; v16bf v; } t;
  const unsigned short* row = p + r * stride;
#pragma unroll
  for (int i = 0; i < 8; ++i) {
    const int k = (i < 4 ? 2 * i : 16 + 2 * (i - 4)) + half * 8;
    t.u[i] = *(const unsigned int*)(row + k);
  }
  return t.v;
}

__device__ __forceinline__ v16bf frag_b(const unsigned short* p, int stride) {
  const int lane = threadIdx.x & 31;
  const int half = lane >> 4, n = lane & 15;
  union { unsigned int u[8]; v16bf v; } t;
  const unsigned short* row = p + n * stride;
#pragma unroll
  for (int i = 0; i < 8; ++i) {
    const int k = (i < 4 ? 2 * i : 16 + 2 * (i - 4)) + half * 8;
    t.u[i] = *(const unsigned int*)(row + k);
  }
  return t.v;
}

__device__ __forceinline__ v8f wmma_bf16(v16bf a, v16bf b, v8f c) {
  return __builtin_amdgcn_wmma_f32_16x16x32_bf16(false, a, false, b,
                                                 (short)0, c, false, false);
}

// ---------------------------------------------------------------------------
__global__ __launch_bounds__(256)
void cvt_bf16_kernel(const float* __restrict__ x, unsigned short* __restrict__ y, int n) {
  int i = blockIdx.x * blockDim.x + threadIdx.x;
  if (i < n) y[i] = f2bf(x[i]);
}

// ---------------------------------------------------------------------------
// LayerNorm: one wave per row of 512, output bf16
// ---------------------------------------------------------------------------
__global__ __launch_bounds__(256)
void ln_kernel(const float* __restrict__ x, const float* __restrict__ w,
               const float* __restrict__ b, unsigned short* __restrict__ y,
               int nrows) {
  const int wid  = (blockIdx.x * blockDim.x + threadIdx.x) >> 5;
  const int lane = threadIdx.x & 31;
  if (wid >= nrows) return;
  const float* row = x + (size_t)wid * DMODEL;

  float vals[16];
  float s = 0.f;
#pragma unroll
  for (int i = 0; i < 16; ++i) { vals[i] = row[lane + i * 32]; s += vals[i]; }
#pragma unroll
  for (int off = 16; off > 0; off >>= 1) s += __shfl_xor(s, off, 32);
  const float mu = s * (1.0f / DMODEL);

  float vs = 0.f;
#pragma unroll
  for (int i = 0; i < 16; ++i) { float d = vals[i] - mu; vs += d * d; }
#pragma unroll
  for (int off = 16; off > 0; off >>= 1) vs += __shfl_xor(vs, off, 32);
  const float rstd = __frsqrt_rn(vs * (1.0f / DMODEL) + 1e-5f);

  unsigned short* orow = y + (size_t)wid * DMODEL;
#pragma unroll
  for (int i = 0; i < 16; ++i) {
    const int idx = lane + i * 32;
    orow[idx] = f2bf((vals[i] - mu) * rstd * w[idx] + b[idx]);
  }
}

// ---------------------------------------------------------------------------
// GEMM: C[M,512] = A[M,512](bf16) * W[512,512]^T(bf16) + bias
// block tile 128x64, 8 waves of 32x32, K-step 32, TDM double-buffered.
// ---------------------------------------------------------------------------
__global__ __launch_bounds__(256)
void proj_kernel(const unsigned short* __restrict__ A,
                 const unsigned short* __restrict__ W,
                 const float* __restrict__ bias,
                 unsigned short* __restrict__ outb,
                 float* __restrict__ outf) {
  __shared__ unsigned short lsA[2][128 * 40];
  __shared__ unsigned short lsB[2][64 * 40];
  const int tid  = threadIdx.x;
  const int wid  = tid >> 5, lane = tid & 31;
  const int half = lane >> 4, ln = lane & 15;
  const int m0 = blockIdx.y * 128;
  const int n0 = blockIdx.x * 64;
  const int wm = wid & 3, wn = wid >> 2;

  v8f acc[2][2] = {};
  int p = 0;

#if HAVE_TDM
  const unsigned long long gA = (unsigned long long)(A + (size_t)m0 * DMODEL);
  const unsigned long long gW = (unsigned long long)(W + (size_t)n0 * DMODEL);
  // prologue: waves 0/1 kick off the first tiles
  if (wid == 0) tdm_load_tile_bf16(gA, lds_addr32(&lsA[0][0]), 32, 128, DMODEL, 3, 3);
  if (wid == 1) tdm_load_tile_bf16(gW, lds_addr32(&lsB[0][0]), 32, 64,  DMODEL, 3, 3);
#endif

  for (int kt = 0; kt < DMODEL; kt += 32) {
#if HAVE_TDM
    if (wid == 0) {
      if (kt + 32 < DMODEL) {
        tdm_load_tile_bf16(gA + (size_t)(kt + 32) * 2, lds_addr32(&lsA[p ^ 1][0]),
                           32, 128, DMODEL, 3, 3);
        __builtin_amdgcn_s_wait_tensorcnt(1);
      } else {
        __builtin_amdgcn_s_wait_tensorcnt(0);
      }
    }
    if (wid == 1) {
      if (kt + 32 < DMODEL) {
        tdm_load_tile_bf16(gW + (size_t)(kt + 32) * 2, lds_addr32(&lsB[p ^ 1][0]),
                           32, 64, DMODEL, 3, 3);
        __builtin_amdgcn_s_wait_tensorcnt(1);
      } else {
        __builtin_amdgcn_s_wait_tensorcnt(0);
      }
    }
    __syncthreads();
#else
#pragma unroll
    for (int i = 0; i < 8; ++i) {            // A tile: 128x32 = 2048 dwords
      const int pi = tid + i * 256;
      const int r = pi >> 4, c2 = pi & 15;
      *(unsigned int*)&lsA[0][r * 40 + c2 * 2] =
          *(const unsigned int*)&A[(size_t)(m0 + r) * DMODEL + kt + c2 * 2];
    }
#pragma unroll
    for (int i = 0; i < 4; ++i) {            // W tile: 64x32 = 1024 dwords
      const int pi = tid + i * 256;
      const int r = pi >> 4, c2 = pi & 15;
      *(unsigned int*)&lsB[0][r * 40 + c2 * 2] =
          *(const unsigned int*)&W[(size_t)(n0 + r) * DMODEL + kt + c2 * 2];
    }
    __syncthreads();
#endif
    const v16bf a0 = frag_a(&lsA[p][(wm * 32 + 0)  * 40], 40);
    const v16bf a1 = frag_a(&lsA[p][(wm * 32 + 16) * 40], 40);
    const v16bf b0 = frag_b(&lsB[p][(wn * 32 + 0)  * 40], 40);
    const v16bf b1 = frag_b(&lsB[p][(wn * 32 + 16) * 40], 40);
    acc[0][0] = wmma_bf16(a0, b0, acc[0][0]);
    acc[0][1] = wmma_bf16(a0, b1, acc[0][1]);
    acc[1][0] = wmma_bf16(a1, b0, acc[1][0]);
    acc[1][1] = wmma_bf16(a1, b1, acc[1][1]);
    __syncthreads();
#if HAVE_TDM
    p ^= 1;
#endif
  }
#pragma unroll
  for (int mi = 0; mi < 2; ++mi)
#pragma unroll
    for (int ni = 0; ni < 2; ++ni) {
      const int n = n0 + wn * 32 + ni * 16 + ln;
      const float bv = bias[n];
#pragma unroll
      for (int r = 0; r < 8; ++r) {
        const int m = m0 + wm * 32 + mi * 16 + half * 8 + r;
        const float v = acc[mi][ni][r] + bv;
        if (outf) outf[(size_t)m * DMODEL + n] = v;
        else      outb[(size_t)m * DMODEL + n] = f2bf(v);
      }
    }
}

// ---------------------------------------------------------------------------
// scores[b,h,q,k] = (qh . kh)/8 + pos_k[q,k], masked -> attn buffer (f32)
// block tile 128(q) x 128(k); K = DK = 64 staged once via TDM.
// ---------------------------------------------------------------------------
__global__ __launch_bounds__(256)
void scores_kernel(const unsigned short* __restrict__ qp,
                   const unsigned short* __restrict__ kp,
                   const float* __restrict__ pos_k,
                   const int* __restrict__ mask,
                   float* __restrict__ attn) {
  __shared__ unsigned short lsQ[128 * 72];
  __shared__ unsigned short lsK[128 * 72];
  const int tid = threadIdx.x, wid = tid >> 5, lane = tid & 31;
  const int half = lane >> 4, ln = lane & 15;
  const int bh = blockIdx.z;
  const int b = bh >> 3, h = bh & 7;
  const int q0 = blockIdx.y * 128;
  const int k0 = blockIdx.x * 128;
  const int wm = wid & 3, wn = wid >> 2;
  const size_t baseQ = ((size_t)b * SEQ + q0) * DMODEL + h * DK;
  const size_t baseK = ((size_t)b * SEQ + k0) * DMODEL + h * DK;

#if HAVE_TDM
  if (wid == 0) {
    tdm_load_tile_bf16((unsigned long long)(qp + baseQ), lds_addr32(&lsQ[0]),
                       64, 128, DMODEL, 4, 3);
    __builtin_amdgcn_s_wait_tensorcnt(0);
  }
  if (wid == 1) {
    tdm_load_tile_bf16((unsigned long long)(kp + baseK), lds_addr32(&lsK[0]),
                       64, 128, DMODEL, 4, 3);
    __builtin_amdgcn_s_wait_tensorcnt(0);
  }
#else
#pragma unroll
  for (int i = 0; i < 16; ++i) {             // 128x64 bf16 = 4096 dwords each
    const int pi = tid + i * 256;
    const int r = pi >> 5, c2 = pi & 31;
    *(unsigned int*)&lsQ[r * 72 + c2 * 2] =
        *(const unsigned int*)&qp[baseQ + (size_t)r * DMODEL + c2 * 2];
    *(unsigned int*)&lsK[r * 72 + c2 * 2] =
        *(const unsigned int*)&kp[baseK + (size_t)r * DMODEL + c2 * 2];
  }
#endif
  __syncthreads();

  v8f acc[2][4] = {};
#pragma unroll
  for (int kf = 0; kf < DK; kf += 32) {
    const v16bf a0 = frag_a(&lsQ[(wm * 32 + 0)  * 72 + kf], 72);
    const v16bf a1 = frag_a(&lsQ[(wm * 32 + 16) * 72 + kf], 72);
#pragma unroll
    for (int ni = 0; ni < 4; ++ni) {
      const v16bf bb = frag_b(&lsK[(wn * 64 + ni * 16) * 72 + kf], 72);
      acc[0][ni] = wmma_bf16(a0, bb, acc[0][ni]);
      acc[1][ni] = wmma_bf16(a1, bb, acc[1][ni]);
    }
  }

  const size_t arow = (size_t)bh * SEQ;
#pragma unroll
  for (int mi = 0; mi < 2; ++mi)
#pragma unroll
    for (int ni = 0; ni < 4; ++ni) {
      const int kk = k0 + wn * 64 + ni * 16 + ln;
#pragma unroll
      for (int r = 0; r < 8; ++r) {
        const int q = q0 + wm * 32 + mi * 16 + half * 8 + r;
        float v = acc[mi][ni][r] * 0.125f + pos_k[(size_t)q * SEQ + kk];
        if (mask[(size_t)q * SEQ + kk] == 0) v = -__builtin_inff();
        attn[(arow + q) * SEQ + kk] = v;
      }
    }
}

// ---------------------------------------------------------------------------
// row softmax over 2048 (one block of 256 per row), in place on attn (f32)
// ---------------------------------------------------------------------------
__global__ __launch_bounds__(256)
void softmax_kernel(float* __restrict__ attn) {
  __shared__ float red[8];
  const int tid = threadIdx.x;
  float* p = attn + (size_t)blockIdx.x * SEQ;

  float v[8];
  float m = -__builtin_inff();
#pragma unroll
  for (int i = 0; i < 8; ++i) { v[i] = p[tid + i * 256]; m = fmaxf(m, v[i]); }
#pragma unroll
  for (int off = 16; off > 0; off >>= 1) m = fmaxf(m, __shfl_xor(m, off, 32));
  if ((tid & 31) == 0) red[tid >> 5] = m;
  __syncthreads();
#pragma unroll
  for (int i = 0; i < 8; ++i) m = fmaxf(m, red[i]);

  float s = 0.f;
#pragma unroll
  for (int i = 0; i < 8; ++i) { v[i] = __expf(v[i] - m); s += v[i]; }
#pragma unroll
  for (int off = 16; off > 0; off >>= 1) s += __shfl_xor(s, off, 32);
  __syncthreads();
  if ((tid & 31) == 0) red[tid >> 5] = s;
  __syncthreads();
  s = 0.f;
#pragma unroll
  for (int i = 0; i < 8; ++i) s += red[i];
  const float inv = 1.0f / s;
#pragma unroll
  for (int i = 0; i < 8; ++i) p[tid + i * 256] = v[i] * inv;
}

// ---------------------------------------------------------------------------
// x[b,q,h*64+dk] = sum_k attn[b,h,q,k] * vh[b,h,k,dk]   (bf16 WMMA)
// attn converted f32->bf16 per tile; V staged transposed as [N][K];
// manual staging (conversion+transpose are not TDM-able) + prefetch.
// ---------------------------------------------------------------------------
__global__ __launch_bounds__(256)
void av_kernel(const float* __restrict__ attn,
               const unsigned short* __restrict__ vp,
               unsigned short* __restrict__ xb) {
  __shared__ unsigned short lsA[128 * 40];
  __shared__ unsigned short lsV[64 * 40];
  const int tid = threadIdx.x, wid = tid >> 5, lane = tid & 31;
  const int half = lane >> 4, ln = lane & 15;
  const int bh = blockIdx.y, b = bh >> 3, h = bh & 7;
  const int q0 = blockIdx.x * 128;
  const int wm = wid & 3, wn = wid >> 2;
  const size_t arow = (size_t)bh * SEQ + q0;

  v8f acc[2][2] = {};
  for (int kt = 0; kt < SEQ; kt += 32) {
#pragma unroll
    for (int i = 0; i < 16; ++i) {           // attn tile 128x32 f32 -> bf16
      const int pi = tid + i * 256;
      const int r = pi >> 5, c = pi & 31;
      lsA[r * 40 + c] = f2bf(attn[(arow + r) * SEQ + kt + c]);
      if (kt + 32 < SEQ)                     // global_prefetch_b8 for next tile
        __builtin_prefetch(&attn[(arow + r) * SEQ + kt + 32 + c], 0, 0);
    }
#pragma unroll
    for (int i = 0; i < 8; ++i) {            // V tile 32(K)x64(N) -> [N][K]
      const int pi = tid + i * 256;
      const int k = pi >> 6, n = pi & 63;
      lsV[n * 40 + k] = vp[((size_t)b * SEQ + kt + k) * DMODEL + h * DK + n];
    }
    __syncthreads();
    const v16bf a0 = frag_a(&lsA[(wm * 32 + 0)  * 40], 40);
    const v16bf a1 = frag_a(&lsA[(wm * 32 + 16) * 40], 40);
    const v16bf b0 = frag_b(&lsV[(wn * 32 + 0)  * 40], 40);
    const v16bf b1 = frag_b(&lsV[(wn * 32 + 16) * 40], 40);
    acc[0][0] = wmma_bf16(a0, b0, acc[0][0]);
    acc[0][1] = wmma_bf16(a0, b1, acc[0][1]);
    acc[1][0] = wmma_bf16(a1, b0, acc[1][0]);
    acc[1][1] = wmma_bf16(a1, b1, acc[1][1]);
    __syncthreads();
  }
#pragma unroll
  for (int mi = 0; mi < 2; ++mi)
#pragma unroll
    for (int ni = 0; ni < 2; ++ni) {
      const int dk = wn * 32 + ni * 16 + ln;
#pragma unroll
      for (int r = 0; r < 8; ++r) {
        const int q = q0 + wm * 32 + mi * 16 + half * 8 + r;
        xb[((size_t)b * SEQ + q) * DMODEL + h * DK + dk] = f2bf(acc[mi][ni][r]);
      }
    }
}

// ---------------------------------------------------------------------------
extern "C" void kernel_launch(void* const* d_in, const int* in_sizes, int n_in,
                              void* d_out, int out_size, void* d_ws, size_t ws_size,
                              hipStream_t stream) {
  const float* q     = (const float*)d_in[0];
  const float* k     = (const float*)d_in[1];
  const float* v     = (const float*)d_in[2];
  const int*   mask  = (const int*)d_in[3];
  const float* pos_k = (const float*)d_in[4];
  const float* ln_w  = (const float*)d_in[5];
  const float* ln_b  = (const float*)d_in[6];
  const float* Wq = (const float*)d_in[7];  const float* bq = (const float*)d_in[8];
  const float* Wk = (const float*)d_in[9];  const float* bk = (const float*)d_in[10];
  const float* Wv = (const float*)d_in[11]; const float* bv = (const float*)d_in[12];
  const float* Wo = (const float*)d_in[13]; const float* bo = (const float*)d_in[14];

  float* out  = (float*)d_out;                              // [B,S,D]
  float* attn = out + (size_t)BATCH * SEQ * DMODEL;         // [B,H,S,S]

  const size_t ROWS = (size_t)BATCH * SEQ;                  // 8192
  unsigned char* ws = (unsigned char*)d_ws;
  size_t off = 0;
  auto alloc = [&](size_t bytes) {
    void* p = ws + off;
    off += (bytes + 255) & ~(size_t)255;
    return p;
  };
  unsigned short* qn  = (unsigned short*)alloc(ROWS * DMODEL * 2);
  unsigned short* kn  = (unsigned short*)alloc(ROWS * DMODEL * 2);
  unsigned short* vn  = (unsigned short*)alloc(ROWS * DMODEL * 2);
  unsigned short* qp  = (unsigned short*)alloc(ROWS * DMODEL * 2);
  unsigned short* kp  = (unsigned short*)alloc(ROWS * DMODEL * 2);
  unsigned short* vp  = (unsigned short*)alloc(ROWS * DMODEL * 2);
  unsigned short* xb  = (unsigned short*)alloc(ROWS * DMODEL * 2);
  unsigned short* Wqb = (unsigned short*)alloc((size_t)DMODEL * DMODEL * 2);
  unsigned short* Wkb = (unsigned short*)alloc((size_t)DMODEL * DMODEL * 2);
  unsigned short* Wvb = (unsigned short*)alloc((size_t)DMODEL * DMODEL * 2);
  unsigned short* Wob = (unsigned short*)alloc((size_t)DMODEL * DMODEL * 2);

  const int wn_elems = DMODEL * DMODEL;                     // 262144
  cvt_bf16_kernel<<<wn_elems / 256, 256, 0, stream>>>(Wq, Wqb, wn_elems);
  cvt_bf16_kernel<<<wn_elems / 256, 256, 0, stream>>>(Wk, Wkb, wn_elems);
  cvt_bf16_kernel<<<wn_elems / 256, 256, 0, stream>>>(Wv, Wvb, wn_elems);
  cvt_bf16_kernel<<<wn_elems / 256, 256, 0, stream>>>(Wo, Wob, wn_elems);

  ln_kernel<<<(int)(ROWS / 8), 256, 0, stream>>>(q, ln_w, ln_b, qn, (int)ROWS);
  ln_kernel<<<(int)(ROWS / 8), 256, 0, stream>>>(k, ln_w, ln_b, kn, (int)ROWS);
  ln_kernel<<<(int)(ROWS / 8), 256, 0, stream>>>(v, ln_w, ln_b, vn, (int)ROWS);

  dim3 pgrid(DMODEL / 64, (unsigned)(ROWS / 128));          // (8, 64)
  proj_kernel<<<pgrid, 256, 0, stream>>>(qn, Wqb, bq, qp, nullptr);
  proj_kernel<<<pgrid, 256, 0, stream>>>(kn, Wkb, bk, kp, nullptr);
  proj_kernel<<<pgrid, 256, 0, stream>>>(vn, Wvb, bv, vp, nullptr);

  dim3 sgrid(SEQ / 128, SEQ / 128, BATCH * NHEAD);          // (16, 16, 32)
  scores_kernel<<<sgrid, 256, 0, stream>>>(qp, kp, pos_k, mask, attn);

  softmax_kernel<<<BATCH * NHEAD * SEQ, 256, 0, stream>>>(attn);

  dim3 agrid(SEQ / 128, BATCH * NHEAD);                     // (16, 32)
  av_kernel<<<agrid, 256, 0, stream>>>(attn, vp, xb);

  proj_kernel<<<pgrid, 256, 0, stream>>>(xb, Wob, bo, nullptr, out);
}
